// NeuroSAT_46170898432684
// MI455X (gfx1250) — matrix operand order repaired and checked by reading, not compiled, not run on hardware
//
#include <hip/hip_runtime.h>
#include <hip/hip_bf16.h>

#define DIM 128
#define N_VARS 50000
#define N_LITS 100000
#define N_CLAUSES 210000
#define N_CELLS 630000
#define N_ROUNDS 8

typedef __attribute__((ext_vector_type(16))) __bf16 v16bf;
typedef __attribute__((ext_vector_type(8)))  float  v8f;

__device__ inline float sigm(float x) { return 1.f / (1.f + __expf(-x)); }

// ---------------------------------------------------------------------------
// small utility kernels
// ---------------------------------------------------------------------------
__global__ void f32_to_bf16_k(const float* __restrict__ src, __bf16* __restrict__ dst, int n) {
    int i = blockIdx.x * blockDim.x + threadIdx.x;
    if (i < n) dst[i] = (__bf16)src[i];
}

__global__ void add_vec_k(const float* __restrict__ a, const float* __restrict__ b,
                          float* __restrict__ d, int n) {
    int i = blockIdx.x * blockDim.x + threadIdx.x;
    if (i < n) d[i] = a[i] + b[i];
}

__global__ void bcast_init_k(const float* __restrict__ wv, const float* __restrict__ bv,
                             float* __restrict__ h, float* __restrict__ c, int total) {
    int i = blockIdx.x * blockDim.x + threadIdx.x;
    if (i >= total) return;
    int j = i & (DIM - 1);
    h[i] = wv[j] + bv[j];
    c[i] = 0.f;
}

// ---------------------------------------------------------------------------
// WMMA fragment helpers (A: 16x32 bf16, lane&15 = M, lane>>4 = K-half;
// B: lane&15 = N-col -> row of W[N,K], 16 contiguous K per lane)
// ---------------------------------------------------------------------------
__device__ inline v16bf load_a_frag(const float* xrow, int k0) {
    v16bf a;
#pragma unroll
    for (int i = 0; i < 8; ++i) a[i] = (__bf16)xrow[k0 + i];
#pragma unroll
    for (int i = 0; i < 8; ++i) a[8 + i] = (__bf16)xrow[k0 + 8 + i];
    return a;
}

// ---------------------------------------------------------------------------
// standalone WMMA GEMM for the 128-wide message layers:
// Y[M,128] = bf16(X[M,128]) @ W[128,128]^T + bias
// ---------------------------------------------------------------------------
__global__ void __launch_bounds__(256)
gemm_msg_k(const float* __restrict__ X, const __bf16* __restrict__ W,
           const float* __restrict__ bias, float* __restrict__ Y) {
    const int lane = threadIdx.x & 31;
    const int wave = threadIdx.x >> 5;
    const int row0 = blockIdx.x * 16;
    const int n0   = wave * 16;

    const float*  xrow = X + (size_t)(row0 + (lane & 15)) * DIM + (lane >> 4) * 16;
    const __bf16* wrow = W + (size_t)(n0 + (lane & 15)) * DIM + (lane >> 4) * 16;

    v8f acc = {};
#pragma unroll
    for (int k0 = 0; k0 < DIM; k0 += 32) {
        v16bf a = load_a_frag(xrow, k0);
        v16bf b = *(const v16bf*)(wrow + k0);
        acc = __builtin_amdgcn_wmma_f32_16x16x32_bf16(false, a, false, b, (short)0, acc,
                                                      false, false);
    }
    const int col   = n0 + (lane & 15);
    const int rbase = row0 + (lane >> 4) * 8;
    const float bv  = bias[col];
#pragma unroll
    for (int v = 0; v < 8; ++v) Y[(size_t)(rbase + v) * DIM + col] = acc[v] + bv;
}

// ---------------------------------------------------------------------------
// fused LSTM update:  gates[16x512] = sum of up to 3 WMMA passes + bias,
// staged in LDS, then elementwise LSTM -> Hout/Cst.  Never touches HBM gates.
// Block = 8 waves; wave w owns gate columns [64w, 64w+64) (4 WMMA tiles).
// Pass 2 (X2) supports the "flipped" row remap: row = (row + row_add) % row_mod.
// ---------------------------------------------------------------------------
__device__ inline void mma_pass(const float* X, const __bf16* W, int ldw,
                                int rowA, int n0, int lane,
                                v8f& a0, v8f& a1, v8f& a2, v8f& a3) {
    const float*  xrow = X + (size_t)rowA * DIM + (lane >> 4) * 16;
    const __bf16* w0   = W + (size_t)(n0 + (lane & 15)) * ldw + (lane >> 4) * 16;
    const size_t  step = (size_t)16 * ldw;
#pragma unroll
    for (int k0 = 0; k0 < DIM; k0 += 32) {
        v16bf a  = load_a_frag(xrow, k0);
        v16bf b0 = *(const v16bf*)(w0 + k0);
        v16bf b1 = *(const v16bf*)(w0 + step + k0);
        v16bf b2 = *(const v16bf*)(w0 + 2 * step + k0);
        v16bf b3 = *(const v16bf*)(w0 + 3 * step + k0);
        a0 = __builtin_amdgcn_wmma_f32_16x16x32_bf16(false, a, false, b0, (short)0, a0, false, false);
        a1 = __builtin_amdgcn_wmma_f32_16x16x32_bf16(false, a, false, b1, (short)0, a1, false, false);
        a2 = __builtin_amdgcn_wmma_f32_16x16x32_bf16(false, a, false, b2, (short)0, a2, false, false);
        a3 = __builtin_amdgcn_wmma_f32_16x16x32_bf16(false, a, false, b3, (short)0, a3, false, false);
    }
}

__global__ void __launch_bounds__(256)
lstm_fused_k(const float* X1, const __bf16* __restrict__ W1, int ldw1,
             const float* X2, const __bf16* __restrict__ W2, int ldw2,
             const float* X3, const __bf16* __restrict__ W3, int ldw3,
             const float* __restrict__ bias,
             float* Hout, float* Cst, int row_add, int row_mod) {
    __shared__ float gsm[16][512];
    const int lane = threadIdx.x & 31;
    const int wave = threadIdx.x >> 5;
    const int row0 = blockIdx.x * 16;
    const int n0   = wave * 64;

    v8f a0 = {}, a1 = {}, a2 = {}, a3 = {};

    const int rowA = row0 + (lane & 15);
    mma_pass(X1, W1, ldw1, rowA, n0, lane, a0, a1, a2, a3);
    if (X2) {
        int r2 = rowA;
        if (row_mod > 0) { r2 += row_add; if (r2 >= row_mod) r2 -= row_mod; }
        mma_pass(X2, W2, ldw2, r2, n0, lane, a0, a1, a2, a3);
    }
    if (X3) mma_pass(X3, W3, ldw3, rowA, n0, lane, a0, a1, a2, a3);

    // write gate strip (+bias) to LDS
    const int colB  = n0 + (lane & 15);
    const int rbase = (lane >> 4) * 8;
    v8f* accs[4] = { &a0, &a1, &a2, &a3 };
#pragma unroll
    for (int tile = 0; tile < 4; ++tile) {
        int col = colB + tile * 16;
        float bv = bias[col];
        v8f acc = *accs[tile];
#pragma unroll
        for (int v = 0; v < 8; ++v) gsm[rbase + v][col] = acc[v] + bv;
    }
    __syncthreads();

    // elementwise LSTM over the block's 16x128 units
#pragma unroll
    for (int e = threadIdx.x; e < 16 * DIM; e += 256) {
        int r = e >> 7, u = e & (DIM - 1);
        float gi = sigm(gsm[r][u]);
        float gf = sigm(gsm[r][DIM + u]);
        float gg = tanhf(gsm[r][2 * DIM + u]);
        float go = sigm(gsm[r][3 * DIM + u]);
        size_t idx = (size_t)(row0 + r) * DIM + u;
        float cn = gf * Cst[idx] + gi * gg;
        Cst[idx]  = cn;
        Hout[idx] = go * tanhf(cn);
    }
}

// ---------------------------------------------------------------------------
// fused prob head: scalar[r] = sum_col relu(X@W1^T + b1)[r,col] * w2[col]
// (b2 / softmax / sigmoid applied by the tiny follow-up kernels)
// ---------------------------------------------------------------------------
__global__ void __launch_bounds__(256)
head_fused_k(const float* __restrict__ X, const __bf16* __restrict__ W1,
             const float* __restrict__ b1, const float* __restrict__ w2,
             float* __restrict__ outs) {
    __shared__ float rowsum[16];
    const int lane = threadIdx.x & 31;
    const int wave = threadIdx.x >> 5;
    const int row0 = blockIdx.x * 16;
    const int n0   = wave * 16;
    if (threadIdx.x < 16) rowsum[threadIdx.x] = 0.f;
    __syncthreads();

    const float*  xrow = X + (size_t)(row0 + (lane & 15)) * DIM + (lane >> 4) * 16;
    const __bf16* wrow = W1 + (size_t)(n0 + (lane & 15)) * DIM + (lane >> 4) * 16;
    v8f acc = {};
#pragma unroll
    for (int k0 = 0; k0 < DIM; k0 += 32) {
        v16bf a = load_a_frag(xrow, k0);
        v16bf b = *(const v16bf*)(wrow + k0);
        acc = __builtin_amdgcn_wmma_f32_16x16x32_bf16(false, a, false, b, (short)0, acc,
                                                      false, false);
    }
    const int col   = n0 + (lane & 15);
    const int rbase = (lane >> 4) * 8;
    const float wv = w2[col];
    const float bb = b1[col];
#pragma unroll
    for (int v = 0; v < 8; ++v) {
        float hval = fmaxf(acc[v] + bb, 0.f);
        atomicAdd(&rowsum[rbase + v], hval * wv);   // ds_add_f32
    }
    __syncthreads();
    if (threadIdx.x < 16) outs[row0 + threadIdx.x] = rowsum[threadIdx.x];
}

__global__ void lit_prob_k(const float* __restrict__ s, const float* __restrict__ b2,
                           float* __restrict__ out) {
    int v = blockIdx.x * blockDim.x + threadIdx.x;
    if (v >= N_VARS) return;
    float pos = s[v] + b2[0], neg = s[v + N_VARS] + b2[0];
    float m  = fmaxf(pos, neg);
    float e0 = __expf(pos - m), e1 = __expf(neg - m);
    float inv = 1.f / (e0 + e1);
    out[v]          = e0 * inv;
    out[N_VARS + v] = e1 * inv;
}

__global__ void clause_prob_k(const float* __restrict__ s, const float* __restrict__ b2,
                              float* __restrict__ out, int M) {
    int c = blockIdx.x * blockDim.x + threadIdx.x;
    if (c < M) out[c] = 1.f / (1.f + __expf(-(s[c] + b2[0])));
}

// ---------------------------------------------------------------------------
// segment-sum via fp32 global atomics: one wave per edge, 4 floats per lane
// ---------------------------------------------------------------------------
__global__ void scatter_add_k(const float* __restrict__ src, const int* __restrict__ sidx,
                              const int* __restrict__ didx, float* __restrict__ dst,
                              int n_cells) {
    int t = blockIdx.x * blockDim.x + threadIdx.x;
    int cell = t >> 5;
    int lane = t & 31;
    if (cell >= n_cells) return;
    const float* s = src + (size_t)sidx[cell] * DIM + lane * 4;
    float*       d = dst + (size_t)didx[cell] * DIM + lane * 4;
    float4 v = *(const float4*)s;
    unsafeAtomicAdd(d + 0, v.x);
    unsafeAtomicAdd(d + 1, v.y);
    unsafeAtomicAdd(d + 2, v.z);
    unsafeAtomicAdd(d + 3, v.w);
}

// ---------------------------------------------------------------------------
extern "C" void kernel_launch(void* const* d_in, const int* in_sizes, int n_in,
                              void* d_out, int out_size, void* d_ws, size_t ws_size,
                              hipStream_t stream) {
    const float* L_init_w = (const float*)d_in[0];
    const float* L_init_b = (const float*)d_in[1];
    const float* C_init_w = (const float*)d_in[2];
    const float* C_init_b = (const float*)d_in[3];
    const float* L_msg_w  = (const float*)d_in[4];
    const float* L_msg_b  = (const float*)d_in[5];
    const float* C_msg_w  = (const float*)d_in[6];
    const float* C_msg_b  = (const float*)d_in[7];
    const float* Cu_wih   = (const float*)d_in[8];
    const float* Cu_bih   = (const float*)d_in[9];
    const float* Cu_whh   = (const float*)d_in[10];
    const float* Cu_bhh   = (const float*)d_in[11];
    const float* Lu_wih   = (const float*)d_in[12];
    const float* Lu_bih   = (const float*)d_in[13];
    const float* Lu_whh   = (const float*)d_in[14];
    const float* Lu_bhh   = (const float*)d_in[15];
    const float* Lp_w1    = (const float*)d_in[16];
    const float* Lp_b1    = (const float*)d_in[17];
    const float* Lp_w2    = (const float*)d_in[18];
    const float* Lp_b2    = (const float*)d_in[19];
    const float* Cp_w1    = (const float*)d_in[20];
    const float* Cp_b1    = (const float*)d_in[21];
    const float* Cp_w2    = (const float*)d_in[22];
    const float* Cp_b2    = (const float*)d_in[23];
    const int*   lit_idx    = (const int*)d_in[24];
    const int*   clause_idx = (const int*)d_in[25];
    float* out = (float*)d_out;

    char* wp = (char*)d_ws;
    auto alloc = [&](size_t bytes) -> char* {
        char* p = wp;
        wp += (bytes + 255) & ~(size_t)255;
        return p;
    };
    float* L_hA  = (float*)alloc((size_t)N_LITS * DIM * 4);
    float* L_hB  = (float*)alloc((size_t)N_LITS * DIM * 4);
    float* L_c   = (float*)alloc((size_t)N_LITS * DIM * 4);
    float* C_h   = (float*)alloc((size_t)N_CLAUSES * DIM * 4);
    float* C_c   = (float*)alloc((size_t)N_CLAUSES * DIM * 4);
    float* tmpL  = (float*)alloc((size_t)N_LITS * DIM * 4);
    float* tmpC  = (float*)alloc((size_t)N_CLAUSES * DIM * 4);
    float* scalL = (float*)alloc((size_t)N_LITS * 4);
    float* scalC = (float*)alloc((size_t)N_CLAUSES * 4);
    __bf16* Lmsg_bf = (__bf16*)alloc((size_t)DIM * DIM * 2);
    __bf16* Cmsg_bf = (__bf16*)alloc((size_t)DIM * DIM * 2);
    __bf16* Cwih_bf = (__bf16*)alloc((size_t)512 * DIM * 2);
    __bf16* Cwhh_bf = (__bf16*)alloc((size_t)512 * DIM * 2);
    __bf16* Lwih_bf = (__bf16*)alloc((size_t)512 * 2 * DIM * 2);
    __bf16* Lwhh_bf = (__bf16*)alloc((size_t)512 * DIM * 2);
    __bf16* Lp1_bf  = (__bf16*)alloc((size_t)DIM * DIM * 2);
    __bf16* Cp1_bf  = (__bf16*)alloc((size_t)DIM * DIM * 2);
    float* bias_C = (float*)alloc(512 * 4);
    float* bias_L = (float*)alloc(512 * 4);

    auto cvt = [&](const float* s, __bf16* d, int n) {
        f32_to_bf16_k<<<(n + 255) / 256, 256, 0, stream>>>(s, d, n);
    };
    cvt(L_msg_w, Lmsg_bf, DIM * DIM);
    cvt(C_msg_w, Cmsg_bf, DIM * DIM);
    cvt(Cu_wih, Cwih_bf, 512 * DIM);
    cvt(Cu_whh, Cwhh_bf, 512 * DIM);
    cvt(Lu_wih, Lwih_bf, 512 * 2 * DIM);
    cvt(Lu_whh, Lwhh_bf, 512 * DIM);
    cvt(Lp_w1, Lp1_bf, DIM * DIM);
    cvt(Cp_w1, Cp1_bf, DIM * DIM);
    add_vec_k<<<2, 256, 0, stream>>>(Cu_bih, Cu_bhh, bias_C, 512);
    add_vec_k<<<2, 256, 0, stream>>>(Lu_bih, Lu_bhh, bias_L, 512);

    bcast_init_k<<<(N_LITS * DIM + 255) / 256, 256, 0, stream>>>(L_init_w, L_init_b, L_hA, L_c,
                                                                 N_LITS * DIM);
    bcast_init_k<<<(N_CLAUSES * DIM + 255) / 256, 256, 0, stream>>>(C_init_w, C_init_b, C_h, C_c,
                                                                    N_CLAUSES * DIM);

    float* L_cur = L_hA;
    float* L_nxt = L_hB;

    for (int t = 0; t < N_ROUNDS; ++t) {
        // ---- literal -> clause messages ----
        gemm_msg_k<<<N_LITS / 16, 256, 0, stream>>>(L_cur, Lmsg_bf, L_msg_b, tmpL);
        hipMemsetAsync(tmpC, 0, (size_t)N_CLAUSES * DIM * 4, stream);
        scatter_add_k<<<(N_CELLS * 32 + 255) / 256, 256, 0, stream>>>(tmpL, lit_idx, clause_idx,
                                                                      tmpC, N_CELLS);
        // ---- clause LSTM (fused: 2 WMMA passes + LSTM, gates live in LDS only) ----
        lstm_fused_k<<<N_CLAUSES / 16, 256, 0, stream>>>(
            tmpC, Cwih_bf, DIM, C_h, Cwhh_bf, DIM, nullptr, nullptr, DIM,
            bias_C, C_h, C_c, 0, 0);

        // ---- clause -> literal messages ----
        gemm_msg_k<<<N_CLAUSES / 16, 256, 0, stream>>>(C_h, Cmsg_bf, C_msg_b, tmpC);
        hipMemsetAsync(tmpL, 0, (size_t)N_LITS * DIM * 4, stream);
        scatter_add_k<<<(N_CELLS * 32 + 255) / 256, 256, 0, stream>>>(tmpC, clause_idx, lit_idx,
                                                                      tmpL, N_CELLS);
        // ---- literal LSTM (fused: CL + flipped + hidden passes); ping-pong L_h ----
        lstm_fused_k<<<N_LITS / 16, 256, 0, stream>>>(
            tmpL, Lwih_bf, 2 * DIM, L_cur, Lwih_bf + DIM, 2 * DIM, L_cur, Lwhh_bf, DIM,
            bias_L, L_nxt, L_c, N_VARS, N_LITS);
        { float* tswap = L_cur; L_cur = L_nxt; L_nxt = tswap; }

        // ---- heads (fused MLP + per-row dot; tiny prob kernels into d_out) ----
        head_fused_k<<<N_LITS / 16, 256, 0, stream>>>(L_cur, Lp1_bf, Lp_b1, Lp_w2, scalL);
        lit_prob_k<<<(N_VARS + 255) / 256, 256, 0, stream>>>(scalL, Lp_b2,
                                                             out + (size_t)t * N_LITS);
        head_fused_k<<<N_CLAUSES / 16, 256, 0, stream>>>(C_h, Cp1_bf, Cp_b1, Cp_w2, scalC);
        clause_prob_k<<<(N_CLAUSES + 255) / 256, 256, 0, stream>>>(
            scalC, Cp_b2, out + (size_t)N_ROUNDS * N_LITS + (size_t)t * N_CLAUSES, N_CLAUSES);
    }
    (void)in_sizes; (void)n_in; (void)out_size; (void)ws_size;
}
